// Gemma2AttentionWrapper_50139448213958
// MI455X (gfx1250) — matrix operand order repaired and checked by reading, not compiled, not run on hardware
//
#include <hip/hip_runtime.h>
#include <hip/hip_bf16.h>
#include <cstdint>

// ---------------------------------------------------------------- types
typedef __attribute__((ext_vector_type(16))) _Float16 v16h;
typedef __attribute__((ext_vector_type(8)))  _Float16 h8;
typedef __attribute__((ext_vector_type(8)))  float    v8f;

union Frag16 { v16h v; h8 h[2]; };

static __device__ __forceinline__ v8f zero8() {
    v8f z;
#pragma unroll
    for (int i = 0; i < 8; ++i) z[i] = 0.0f;
    return z;
}

#define WMMA(a, b, c) \
    __builtin_amdgcn_wmma_f32_16x16x32_f16(false, (a), false, (b), (short)0, (c), false, false)

static constexpr int HID  = 2304;
static constexpr int SEQ  = 2048;
static constexpr int HD   = 256;
static constexpr int NH   = 8;
static constexpr int NKV  = 4;
static constexpr int WIN  = 1024;
static constexpr float SOFTCAP = 50.0f;

// ------------------------------------------------- async Global->LDS (CDNA5)
// builtin signature (from clang diagnostic): param0 = int4 __device__* (AS1)
typedef int v4i_t __attribute__((vector_size(16)));
typedef __attribute__((address_space(1))) v4i_t* gas_v4i;
typedef __attribute__((address_space(3))) v4i_t* las_v4i;

static __device__ __forceinline__ void async_copy_b128(const void* g, void* l) {
#if __has_builtin(__builtin_amdgcn_global_load_async_to_lds_b128)
    __builtin_amdgcn_global_load_async_to_lds_b128(
        (gas_v4i)(uintptr_t)g, (las_v4i)(uint32_t)(uintptr_t)l, 0, 0);
#else
    // flat LDS address low 32 bits == LDS byte offset (aperture truncation rule)
    asm volatile("global_load_async_to_lds_b128 %0, %1, off"
                 :: "v"((uint32_t)(uintptr_t)l), "v"(g)
                 : "memory");
#endif
}

static __device__ __forceinline__ void async_wait0() {
#if __has_builtin(__builtin_amdgcn_s_wait_asynccnt)
    __builtin_amdgcn_s_wait_asynccnt(0);
#else
    asm volatile("s_wait_asynccnt 0x0" ::: "memory");
#endif
}

// ---------------------------------------------------------------- f32 -> f16 convert
__global__ void cvt_f32_f16(const float* __restrict__ in, _Float16* __restrict__ out, int n) {
    for (int i = blockIdx.x * 256 + threadIdx.x; i < n; i += gridDim.x * 256)
        out[i] = (_Float16)in[i];
}

// ---------------------------------------------------------------- RMSNorm -> f16
__global__ void rmsnorm_kernel(const float* __restrict__ x, const float* __restrict__ w,
                               _Float16* __restrict__ out) {
    int row = blockIdx.x;
    const float* xr = x + (size_t)row * HID;
    float ss = 0.0f;
    for (int i = threadIdx.x; i < HID; i += 256) { float v = xr[i]; ss += v * v; }
    __shared__ float red[256];
    red[threadIdx.x] = ss;
    __syncthreads();
    for (int s = 128; s > 0; s >>= 1) {
        if (threadIdx.x < s) red[threadIdx.x] += red[threadIdx.x + s];
        __syncthreads();
    }
    float inv = rsqrtf(red[0] / (float)HID + 1e-6f);
    for (int i = threadIdx.x; i < HID; i += 256)
        out[(size_t)row * HID + i] = (_Float16)(xr[i] * inv * (1.0f + w[i]));
}

// ---------------------------------------------------------------- WMMA GEMM
// C[M,N] = A[M,K] (row-major f16) * B[N,K]^T (row-major f16)
// Double-buffered async Global->LDS pipeline, one barrier per K-step.
// mode 0: f32 out     mode 1: f16 out     mode 2: f32 out + residual
__global__ __launch_bounds__(256, 1)
void gemm_f16(const _Float16* __restrict__ A, const _Float16* __restrict__ Bm,
              void* __restrict__ Cout, const float* __restrict__ Res,
              int M, int N, int K, int mode) {
    __shared__ __align__(16) _Float16 As[2][128 * 40];
    __shared__ __align__(16) _Float16 Bs[2][128 * 40];

    const int m0 = blockIdx.y * 128;
    const int n0 = blockIdx.x * 128;
    const int tid  = threadIdx.x;
    const int wave = tid >> 5;
    const int lane = tid & 31;
    const int wm = wave & 3;   // 4 waves along M (32 rows each)
    const int wn = wave >> 2;  // 2 waves along N (64 cols each)

    v8f acc[2][4];
#pragma unroll
    for (int i = 0; i < 2; ++i)
#pragma unroll
        for (int j = 0; j < 4; ++j) acc[i][j] = zero8();

    const int lrow = tid >> 1;          // 0..127
    const int lcol = (tid & 1) * 16;    // 0 or 16 (halves)
    const _Float16* ag = A  + (size_t)(m0 + lrow) * K + lcol;
    const _Float16* bg = Bm + (size_t)(n0 + lrow) * K + lcol;

    auto issue = [&](int st, int k0) {
        async_copy_b128(ag + k0,     &As[st][lrow * 40 + lcol]);
        async_copy_b128(ag + k0 + 8, &As[st][lrow * 40 + lcol + 8]);
        async_copy_b128(bg + k0,     &Bs[st][lrow * 40 + lcol]);
        async_copy_b128(bg + k0 + 8, &Bs[st][lrow * 40 + lcol + 8]);
    };

    issue(0, 0);
    int stage = 0;
    for (int k0 = 0; k0 < K; k0 += 32, stage ^= 1) {
        async_wait0();        // my stage-k copies landed
        __syncthreads();      // everyone's landed; everyone done reading stage k-1
        if (k0 + 32 < K) issue(stage ^ 1, k0 + 32);  // overlap next copies w/ compute

        Frag16 af[2], bf[4];
        const int klo  = (lane < 16) ? 0 : 8;    // A: K halves {klo..+7, klo+16..+23}
        const int bklo = (lane < 16) ? 0 : 8;    // B: K halves {bklo..+7, bklo+16..+23}
#pragma unroll
        for (int i = 0; i < 2; ++i) {
            const _Float16* p = &As[stage][(wm * 32 + i * 16 + (lane & 15)) * 40];
            af[i].h[0] = *(const h8*)(p + klo);
            af[i].h[1] = *(const h8*)(p + klo + 16);
        }
#pragma unroll
        for (int j = 0; j < 4; ++j) {
            const _Float16* p = &Bs[stage][(wn * 64 + j * 16 + (lane & 15)) * 40 + bklo];
            bf[j].h[0] = *(const h8*)(p);
            bf[j].h[1] = *(const h8*)(p + 16);
        }
#pragma unroll
        for (int i = 0; i < 2; ++i)
#pragma unroll
            for (int j = 0; j < 4; ++j)
                acc[i][j] = WMMA(af[i].v, bf[j].v, acc[i][j]);
    }

    const int mo = (lane < 16) ? 0 : 8;
    const int nc = lane & 15;
#pragma unroll
    for (int i = 0; i < 2; ++i)
#pragma unroll
        for (int j = 0; j < 4; ++j)
#pragma unroll
            for (int r = 0; r < 8; ++r) {
                int row = m0 + wm * 32 + i * 16 + r + mo;
                int col = n0 + wn * 64 + j * 16 + nc;
                size_t idx = (size_t)row * N + col;
                float v = acc[i][j][r];
                if (mode == 0)       ((float*)Cout)[idx] = v;
                else if (mode == 1)  ((_Float16*)Cout)[idx] = (_Float16)v;
                else                 ((float*)Cout)[idx] = v + Res[idx];
            }
}

// ---------------------------------------------------------------- RoPE + layout permutes
// role 0..7  : q head, rope, *1/16 scale -> q16 [b,h,s,d]
// role 8..11 : k head, rope              -> k16 [b,kv,s,d]
// role 12..15: v head, transpose         -> v16t [b,kv,d,s]
__global__ void rope_kernel(const float* __restrict__ qf, const float* __restrict__ kf,
                            const float* __restrict__ vf,
                            const float* __restrict__ cosb, const float* __restrict__ sinb,
                            _Float16* __restrict__ q16, _Float16* __restrict__ k16,
                            _Float16* __restrict__ v16t) {
    int bs = blockIdx.x;              // b*SEQ + s
    int role = blockIdx.y;            // 0..15
    int b = bs / SEQ, s = bs % SEQ;
    int t = threadIdx.x;              // 0..127
    size_t cb = (size_t)bs * HD;
    if (role < NH) {
        const float* src = qf + (size_t)bs * (NH * HD) + role * HD;
        float x1 = src[t], x2 = src[t + 128];
        float c1 = cosb[cb + t],       s1 = sinb[cb + t];
        float c2 = cosb[cb + t + 128], s2 = sinb[cb + t + 128];
        _Float16* dst = q16 + (((size_t)b * NH + role) * SEQ + s) * HD;
        dst[t]       = (_Float16)((x1 * c1 - x2 * s1) * 0.0625f);  // fold head_dim^-0.5
        dst[t + 128] = (_Float16)((x2 * c2 + x1 * s2) * 0.0625f);
    } else if (role < NH + NKV) {
        int kv = role - NH;
        const float* src = kf + (size_t)bs * (NKV * HD) + kv * HD;
        float x1 = src[t], x2 = src[t + 128];
        float c1 = cosb[cb + t],       s1 = sinb[cb + t];
        float c2 = cosb[cb + t + 128], s2 = sinb[cb + t + 128];
        _Float16* dst = k16 + (((size_t)b * NKV + kv) * SEQ + s) * HD;
        dst[t]       = (_Float16)(x1 * c1 - x2 * s1);
        dst[t + 128] = (_Float16)(x2 * c2 + x1 * s2);
    } else {
        int kv = role - NH - NKV;
        const float* src = vf + (size_t)bs * (NKV * HD) + kv * HD;
        _Float16* dst = v16t + ((size_t)b * NKV + kv) * HD * SEQ;
        dst[(size_t)t * SEQ + s]         = (_Float16)src[t];
        dst[(size_t)(t + 128) * SEQ + s] = (_Float16)src[t + 128];
    }
}

// ---------------------------------------------------------------- flash attention (WMMA)
// grid (SEQ/128, NH, B), 256 threads = 8 independent waves, 16 q-rows each.
__global__ __launch_bounds__(256, 1)
void attn_kernel(const _Float16* __restrict__ q16, const _Float16* __restrict__ k16,
                 const _Float16* __restrict__ v16t, _Float16* __restrict__ attn) {
    const int h  = blockIdx.y;
    const int b  = blockIdx.z;
    const int wave = threadIdx.x >> 5;
    const int lane = threadIdx.x & 31;
    const int t0 = blockIdx.x * 128 + wave * 16;     // first q row of this wave
    const int kv = h >> 1;                            // GQA groups=2

    const _Float16* qbase = q16  + ((size_t)b * NH  + h ) * SEQ * HD;
    const _Float16* kbase = k16  + ((size_t)b * NKV + kv) * SEQ * HD;
    const _Float16* vbase = v16t + ((size_t)b * NKV + kv) * HD * SEQ;

    __shared__ __align__(16) _Float16 pbuf[8][16 * 40];   // per-wave P staging
    _Float16* pl = pbuf[wave];

    // Persistent Q fragments (A layout), d = 0..255 in 8 frags of K=32
    Frag16 qf[8];
    {
        const _Float16* qr = qbase + (size_t)(t0 + (lane & 15)) * HD;
        const int klo = (lane < 16) ? 0 : 8;
#pragma unroll
        for (int f = 0; f < 8; ++f) {
            qf[f].h[0] = *(const h8*)(qr + f * 32 + klo);
            qf[f].h[1] = *(const h8*)(qr + f * 32 + klo + 16);
        }
    }

    v8f acc[16];
#pragma unroll
    for (int i = 0; i < 16; ++i) acc[i] = zero8();
    float m_i[8], l_i[8];
#pragma unroll
    for (int r = 0; r < 8; ++r) { m_i[r] = -1e30f; l_i[r] = 0.0f; }

    int kb_lo = t0 - (WIN - 1);
    if (kb_lo < 0) kb_lo = 0;
    kb_lo &= ~31;
    const int kb_hi = t0 + 16;
    const int mrow = (lane < 16) ? 0 : 8;
    const int ncol = lane & 15;

    for (int kb = kb_lo; kb < kb_hi; kb += 32) {
        // ---- scores S = Q * K^T   (two 16x16 tiles: keys kb..+15, kb+16..+31)
        v8f s0 = zero8(), s1 = zero8();
#pragma unroll
        for (int ks = 0; ks < 8; ++ks) {
            const int dlo = ks * 32 + ((lane < 16) ? 0 : 8);
            const _Float16* kr0 = kbase + (size_t)(kb + ncol) * HD + dlo;
            const _Float16* kr1 = kr0 + 16 * HD;
            Frag16 b0, b1;
            b0.h[0] = *(const h8*)(kr0); b0.h[1] = *(const h8*)(kr0 + 16);
            b1.h[0] = *(const h8*)(kr1); b1.h[1] = *(const h8*)(kr1 + 16);
            s0 = WMMA(qf[ks].v, b0.v, s0);
            s1 = WMMA(qf[ks].v, b1.v, s1);
        }

        // ---- softcap + sliding-window mask + online softmax
        const int key0 = kb + ncol;
        const int key1 = key0 + 16;
        float corr[8];
#pragma unroll
        for (int i = 0; i < 8; ++i) {
            const int qm = t0 + i + mrow;
            float x0 = SOFTCAP * tanhf(s0[i] * (1.0f / SOFTCAP));
            float x1 = SOFTCAP * tanhf(s1[i] * (1.0f / SOFTCAP));
            const bool a0 = (key0 <= qm) && (qm - key0 < WIN);
            const bool a1 = (key1 <= qm) && (qm - key1 < WIN);
            x0 = a0 ? x0 : -1e30f;
            x1 = a1 ? x1 : -1e30f;
            float mx = fmaxf(x0, x1);
#pragma unroll
            for (int off = 8; off >= 1; off >>= 1)
                mx = fmaxf(mx, __shfl_xor(mx, off, 32));   // within 16-lane half
            const float nm = fmaxf(m_i[i], mx);
            corr[i] = expf(m_i[i] - nm);
            const float p0 = a0 ? expf(x0 - nm) : 0.0f;
            const float p1 = a1 ? expf(x1 - nm) : 0.0f;
            s0[i] = p0; s1[i] = p1;
            float rs = p0 + p1;
#pragma unroll
            for (int off = 8; off >= 1; off >>= 1)
                rs += __shfl_xor(rs, off, 32);
            l_i[i] = l_i[i] * corr[i] + rs;
            m_i[i] = nm;
        }
#pragma unroll
        for (int t = 0; t < 16; ++t)
#pragma unroll
            for (int i = 0; i < 8; ++i) acc[t][i] *= corr[i];

        // ---- C layout -> A layout for P via per-wave LDS (same-wave DS is in-order)
#pragma unroll
        for (int i = 0; i < 8; ++i) {
            pl[(i + mrow) * 40 + ncol]      = (_Float16)s0[i];
            pl[(i + mrow) * 40 + ncol + 16] = (_Float16)s1[i];
        }
        Frag16 pf;
        {
            const _Float16* pr = pl + (lane & 15) * 40;
            const int klo = (lane < 16) ? 0 : 8;
            pf.h[0] = *(const h8*)(pr + klo);
            pf.h[1] = *(const h8*)(pr + klo + 16);
        }

        // ---- O += P * V  (V pre-transposed: lane n = column d, elements = keys)
#pragma unroll
        for (int dt = 0; dt < 16; ++dt) {
            const _Float16* vr = vbase + (size_t)(dt * 16 + ncol) * SEQ + kb
                                 + ((lane < 16) ? 0 : 8);
            Frag16 vfr;
            vfr.h[0] = *(const h8*)(vr);
            vfr.h[1] = *(const h8*)(vr + 16);
            acc[dt] = WMMA(pf.v, vfr.v, acc[dt]);
        }
    }

    // ---- normalize + store to attn_f16 [B*S, NH*HD]
#pragma unroll
    for (int i = 0; i < 8; ++i) l_i[i] = 1.0f / l_i[i];
#pragma unroll
    for (int dt = 0; dt < 16; ++dt)
#pragma unroll
        for (int i = 0; i < 8; ++i) {
            const int r = t0 + i + mrow;
            size_t idx = ((size_t)b * SEQ + r) * (NH * HD) + h * HD + dt * 16 + ncol;
            attn[idx] = (_Float16)(acc[dt][i] * l_i[i]);
        }
}

// ---------------------------------------------------------------- launch
extern "C" void kernel_launch(void* const* d_in, const int* in_sizes, int n_in,
                              void* d_out, int out_size, void* d_ws, size_t ws_size,
                              hipStream_t stream) {
    const float* hidden = (const float*)d_in[0];
    const float* cosb   = (const float*)d_in[1];
    const float* sinb   = (const float*)d_in[2];
    const float* rmsw   = (const float*)d_in[3];
    const float* Wq     = (const float*)d_in[4];
    const float* Wk     = (const float*)d_in[5];
    const float* Wv     = (const float*)d_in[6];
    const float* Wo     = (const float*)d_in[7];
    float* out = (float*)d_out;

    const int B = in_sizes[0] / (SEQ * HID);   // 2
    const int M = B * SEQ;                     // 4096

    char* ws = (char*)d_ws;
    auto alloc = [&](size_t bytes) -> char* {
        char* p = ws;
        ws += (bytes + 255) & ~(size_t)255;
        return p;
    };
    _Float16* x16   = (_Float16*)alloc((size_t)M * HID * 2);
    _Float16* wq16  = (_Float16*)alloc((size_t)NH  * HD * HID * 2);
    _Float16* wk16  = (_Float16*)alloc((size_t)NKV * HD * HID * 2);
    _Float16* wv16  = (_Float16*)alloc((size_t)NKV * HD * HID * 2);
    _Float16* wo16  = (_Float16*)alloc((size_t)HID * NH * HD * 2);
    float*    qf32  = (float*)   alloc((size_t)M * NH  * HD * 4);
    float*    kf32  = (float*)   alloc((size_t)M * NKV * HD * 4);
    float*    vf32  = (float*)   alloc((size_t)M * NKV * HD * 4);
    _Float16* q16   = (_Float16*)alloc((size_t)M * NH  * HD * 2);
    _Float16* k16   = (_Float16*)alloc((size_t)M * NKV * HD * 2);
    _Float16* v16t  = (_Float16*)alloc((size_t)M * NKV * HD * 2);
    _Float16* at16  = (_Float16*)alloc((size_t)M * NH  * HD * 2);

    // weights -> f16
    cvt_f32_f16<<<2048, 256, 0, stream>>>(Wq, wq16, NH  * HD * HID);
    cvt_f32_f16<<<2048, 256, 0, stream>>>(Wk, wk16, NKV * HD * HID);
    cvt_f32_f16<<<2048, 256, 0, stream>>>(Wv, wv16, NKV * HD * HID);
    cvt_f32_f16<<<2048, 256, 0, stream>>>(Wo, wo16, HID * NH * HD);

    // RMSNorm
    rmsnorm_kernel<<<M, 256, 0, stream>>>(hidden, rmsw, x16);

    // QKV projections (WMMA, async-LDS pipelined)
    gemm_f16<<<dim3((NH * HD) / 128, M / 128), 256, 0, stream>>>(
        x16, wq16, qf32, nullptr, M, NH * HD, HID, 0);
    gemm_f16<<<dim3((NKV * HD) / 128, M / 128), 256, 0, stream>>>(
        x16, wk16, kf32, nullptr, M, NKV * HD, HID, 0);
    gemm_f16<<<dim3((NKV * HD) / 128, M / 128), 256, 0, stream>>>(
        x16, wv16, vf32, nullptr, M, NKV * HD, HID, 0);

    // RoPE + permutes
    rope_kernel<<<dim3(M, NH + 2 * NKV), 128, 0, stream>>>(
        qf32, kf32, vf32, cosb, sinb, q16, k16, v16t);

    // flash attention (WMMA)
    attn_kernel<<<dim3(SEQ / 128, NH, B), 256, 0, stream>>>(q16, k16, v16t, at16);

    // output projection + residual (WMMA)
    gemm_f16<<<dim3(HID / 128, M / 128), 256, 0, stream>>>(
        at16, wo16, out, hidden, M, HID, NH * HD, 2);
}
